// AxialAttention_30760555774223
// MI455X (gfx1250) — matrix operand order repaired and checked by reading, not compiled, not run on hardware
//
#include <hip/hip_runtime.h>

#define HEADS_N 8
#define DH 64
#define D_FEAT 256
#define INNER 512
#define N_ROWS 128
#define W_TOK 256
#define T_TOK (N_ROWS * W_TOK)   // 32768
#define SCALE_Q 0.125f           // 64^-0.5

typedef __attribute__((ext_vector_type(16))) _Float16 v16h;
typedef __attribute__((ext_vector_type(8)))  _Float16 v8h;
typedef __attribute__((ext_vector_type(8)))  float    v8f;
typedef _Float16 half_t;

union V16U { v16h v; v8h h[2]; };

__device__ __forceinline__ v8f wmma32(v16h a, v16h b, v8f c) {
  // D = A(16x32 f16) * B(32x16 f16) + C(16x16 f32)
  return __builtin_amdgcn_wmma_f32_16x16x32_f16(false, a, false, b, (short)0, c,
                                                false, false);
}

__device__ __forceinline__ v8f zero8() {
  v8f z = {0.f, 0.f, 0.f, 0.f, 0.f, 0.f, 0.f, 0.f};
  return z;
}

// A-operand loader: A row-major (lda elements). ISA 16-bit A 16x32 layout:
// lane<16: M=lane, K = k0+0..7 (VGPR0-3) and k0+16..23 (VGPR4-7)
// lane>=16: M=lane-16, K = k0+8..15 and k0+24..31
template <typename PT>
__device__ __forceinline__ v16h load_a16(PT a, int lda, int m0, int k0, int lane) {
  PT p = a + (m0 + (lane & 15)) * lda + k0 + ((lane >> 4) << 3);
  V16U u;
  u.h[0] = *(const v8h*)(p);
  u.h[1] = *(const v8h*)(p + 16);
  return u.v;
}

// B-operand loader from an N-major ("B transposed") buffer bt[n][k], ld = ldb.
// ISA 16-bit B 32x16 layout: lane%16 = N column; lanes 0-15 hold K 0-7,16-23,
// lanes 16-31 hold K 8-15,24-31 (same K order as the A loader).
template <typename PT>
__device__ __forceinline__ v16h load_bt16(PT bt, int ldb, int n0, int k0, int lane) {
  PT p = bt + (n0 + (lane & 15)) * ldb + k0 + ((lane >> 4) << 3);
  V16U u;
  u.h[0] = *(const v8h*)(p);
  u.h[1] = *(const v8h*)(p + 16);
  return u.v;
}

// ---- CDNA5 async global->LDS copy (ASYNCcnt path, CDNA5 ISA) ----
// LDS destination = byte offset in the vdst VGPR; global addr from 64-bit VGPR pair.
__device__ __forceinline__ void async_copy_b128(unsigned lds_byte, const void* gaddr) {
  asm volatile("global_load_async_to_lds_b128 %0, %1, off"
               :
               : "v"(lds_byte), "v"(gaddr)
               : "memory");
}
__device__ __forceinline__ void wait_async_all() {
  asm volatile("s_wait_asynccnt 0x0" ::: "memory");
}

// Generic LDS pointer -> 32-bit LDS byte offset (aperture truncation, ISA 10.2)
__device__ __forceinline__ unsigned lds_off(const void* p) {
  return (unsigned)(size_t)p;
}

// -------- weight transpose+convert: dst[n*K + k] = (f16)src[k*ld + n] --------
__global__ void cvt_w_kernel(const float* __restrict__ src, int ld, int K, int N,
                             half_t* __restrict__ dst) {
  const int total = K * N;
  for (int i = blockIdx.x * blockDim.x + threadIdx.x; i < total;
       i += gridDim.x * blockDim.x) {
    const int n = i / K, kk = i % K;
    dst[i] = (half_t)src[(size_t)kk * ld + n];
  }
}

// -------- LayerNorm: one wave32 per token, 256 features --------
__global__ __launch_bounds__(256) void ln_kernel(const float* __restrict__ x,
                                                 const float* __restrict__ g,
                                                 const float* __restrict__ b,
                                                 half_t* __restrict__ xn) {
  const int gid = blockIdx.x * blockDim.x + threadIdx.x;
  const int token = gid >> 5;
  const int lane = threadIdx.x & 31;
  if (token >= T_TOK) return;
  const float* xr = x + (size_t)token * D_FEAT;
  float v[8];
  float s = 0.f;
  #pragma unroll
  for (int i = 0; i < 8; ++i) { v[i] = xr[lane * 8 + i]; s += v[i]; }
  #pragma unroll
  for (int m = 1; m < 32; m <<= 1) s += __shfl_xor(s, m, 32);
  const float mu = s * (1.0f / 256.0f);
  float vs = 0.f;
  #pragma unroll
  for (int i = 0; i < 8; ++i) { const float d = v[i] - mu; vs += d * d; }
  #pragma unroll
  for (int m = 1; m < 32; m <<= 1) vs += __shfl_xor(vs, m, 32);
  const float rstd = rsqrtf(vs * (1.0f / 256.0f) + 1e-5f);
  half_t* o = xn + (size_t)token * D_FEAT + lane * 8;
  #pragma unroll
  for (int i = 0; i < 8; ++i)
    o[i] = (half_t)((v[i] - mu) * rstd * g[lane * 8 + i] + b[lane * 8 + i]);
}

// -------- fused projection GEMM: xn(32768x256) @ [Wq|Wk|Wv|Wg](256x2048) --------
// Wave owns a 64x32 C strip (8 WMMA tiles); A operand reused across 2 B tiles.
// q,k written [row][head][token][64]; v written transposed [row][head][64][token];
// gate = sigmoid(.+bg) written [token][512] f16.
__global__ __launch_bounds__(256) void proj_kernel(
    const half_t* __restrict__ xn, const half_t* __restrict__ Wt,
    const float* __restrict__ bg, half_t* __restrict__ qo,
    half_t* __restrict__ ko, half_t* __restrict__ vo, half_t* __restrict__ go) {
  const int tid = threadIdx.x;
  const int wave = tid >> 5, lane = tid & 31;
  const int m_base = blockIdx.x * 128 + (wave >> 2) * 64;
  const int n_base = blockIdx.y * 128 + (wave & 3) * 32;

  v8f c[4][2];
  #pragma unroll
  for (int t = 0; t < 4; ++t) { c[t][0] = zero8(); c[t][1] = zero8(); }

  #pragma unroll
  for (int ks = 0; ks < D_FEAT; ks += 32) {
    const v16h b0 = load_bt16(Wt, D_FEAT, n_base, ks, lane);
    const v16h b1 = load_bt16(Wt, D_FEAT, n_base + 16, ks, lane);
    #pragma unroll
    for (int t = 0; t < 4; ++t) {
      const v16h a = load_a16(xn, D_FEAT, m_base + 16 * t, ks, lane);
      c[t][0] = wmma32(a, b0, c[t][0]);
      c[t][1] = wmma32(a, b1, c[t][1]);
    }
  }

  #pragma unroll
  for (int u = 0; u < 2; ++u) {
    const int n = n_base + u * 16 + (lane & 15);
    #pragma unroll
    for (int t = 0; t < 4; ++t) {
      const float* cf = (const float*)&c[t][u];
      const int mrow = m_base + 16 * t + ((lane >> 4) << 3);
      #pragma unroll
      for (int r = 0; r < 8; ++r) {
        const int token = mrow + r;
        const int row = token >> 8, tt = token & 255;
        const float val = cf[r];
        if (n < 512) {
          const int head = n >> 6, dh = n & 63;
          qo[((size_t)(row * 8 + head) * 256 + tt) * 64 + dh] =
              (half_t)(val * SCALE_Q);
        } else if (n < 1024) {
          const int nc = n - 512, head = nc >> 6, dh = nc & 63;
          ko[((size_t)(row * 8 + head) * 256 + tt) * 64 + dh] = (half_t)val;
        } else if (n < 1536) {
          const int nc = n - 1024, head = nc >> 6, dh = nc & 63;
          vo[((size_t)(row * 8 + head) * 64 + dh) * 256 + tt] = (half_t)val;
        } else {
          const int nc = n - 1536;
          const float sg = 1.0f / (1.0f + __expf(-(val + bg[nc])));
          go[(size_t)token * 512 + nc] = (half_t)sg;
        }
      }
    }
  }
}

// -------- attention: one workgroup per (row, head); 8 waves --------
__global__ __launch_bounds__(256) void attn_kernel(
    const half_t* __restrict__ q, const half_t* __restrict__ k,
    const half_t* __restrict__ vT, const half_t* __restrict__ gate,
    half_t* __restrict__ go) {
  extern __shared__ half_t smem[];
  half_t* kb = smem;             // k  : 256 tokens x 64  (B-operand N-major)
  half_t* vb = smem + 16384;     // vT : 64 dims  x 256   (B-operand N-major)
  half_t* pbase = smem + 32768;  // P  : 8 waves x 16 x 256

  const int tid = threadIdx.x;
  const int wave = tid >> 5, lane = tid & 31;
  const int row = blockIdx.x >> 3, head = blockIdx.x & 7;
  const size_t rh = (size_t)(row * 8 + head);
  const half_t* qg = q + rh * (256 * 64);
  const half_t* kg = k + rh * (256 * 64);
  const half_t* vg = vT + rh * (64 * 256);

  {  // async stage of K and V^T into LDS (32 KB each), ASYNCcnt-tracked
    const unsigned kb_off = lds_off(kb);
    const unsigned vb_off = lds_off(vb);
    for (int i = tid; i < 2048; i += 256) {
      async_copy_b128(kb_off + (unsigned)i * 16u, kg + (size_t)i * 8);
      async_copy_b128(vb_off + (unsigned)i * 16u, vg + (size_t)i * 8);
    }
    wait_async_all();
  }
  __syncthreads();

  half_t* pb = pbase + wave * (16 * 256);
  const int lr = (lane >> 4) << 3;  // row offset inside C tile
  const int lc = lane & 15;         // column inside C tile

  for (int rt = wave * 2; rt < wave * 2 + 2; ++rt) {
    const int m0 = rt * 16;
    const v16h a0 = load_a16(qg, 64, m0, 0, lane);
    const v16h a1 = load_a16(qg, 64, m0, 32, lane);

    v8f sc[16];  // full 16x256 score strip in registers
    #pragma unroll
    for (int j = 0; j < 16; ++j) {
      v8f cc = zero8();
      cc = wmma32(a0, load_bt16(kb, 64, j * 16, 0, lane), cc);
      cc = wmma32(a1, load_bt16(kb, 64, j * 16, 32, lane), cc);
      sc[j] = cc;
    }

    // softmax: each (VGPR r, lane-half) owns one row; columns live across the
    // 16 lanes of the half and across the 16 tiles.
    float sm[8];
    #pragma unroll
    for (int r = 0; r < 8; ++r) {
      float m = ((float*)&sc[0])[r];
      #pragma unroll
      for (int j = 1; j < 16; ++j) m = fmaxf(m, ((float*)&sc[j])[r]);
      m = fmaxf(m, __shfl_xor(m, 1, 32));
      m = fmaxf(m, __shfl_xor(m, 2, 32));
      m = fmaxf(m, __shfl_xor(m, 4, 32));
      m = fmaxf(m, __shfl_xor(m, 8, 32));
      float s = 0.f;
      #pragma unroll
      for (int j = 0; j < 16; ++j) {
        const float e = __expf(((float*)&sc[j])[r] - m);
        ((float*)&sc[j])[r] = e;
        s += e;
      }
      s += __shfl_xor(s, 1, 32);
      s += __shfl_xor(s, 2, 32);
      s += __shfl_xor(s, 4, 32);
      s += __shfl_xor(s, 8, 32);
      sm[r] = 1.0f / s;
    }

    // re-layout P via per-wave LDS buffer (row-major 16 x 256)
    #pragma unroll
    for (int j = 0; j < 16; ++j) {
      #pragma unroll
      for (int r = 0; r < 8; ++r)
        pb[(lr + r) * 256 + j * 16 + lc] = (half_t)(((float*)&sc[j])[r] * sm[r]);
    }

    // out(16x64) = P(16x256) @ V(256x64); gate + store f16
    #pragma unroll
    for (int nt = 0; nt < 4; ++nt) {
      v8f co = zero8();
      #pragma unroll
      for (int ks = 0; ks < 256; ks += 32)
        co = wmma32(load_a16(pb, 256, 0, ks, lane),
                    load_bt16(vb, 256, nt * 16, ks, lane), co);
      const int dh = nt * 16 + lc;
      const int mrow = m0 + lr;
      const float* cf = (const float*)&co;
      #pragma unroll
      for (int r = 0; r < 8; ++r) {
        const size_t token = (size_t)row * 256 + mrow + r;
        const int col = head * 64 + dh;
        const float gv = (float)gate[token * 512 + col];
        go[token * 512 + col] = (half_t)(cf[r] * gv);
      }
    }
  }
}

// -------- output GEMM: go(32768x512) @ Wo(512x256) + bo -> fp32 --------
__global__ __launch_bounds__(256) void outproj_kernel(
    const half_t* __restrict__ go, const half_t* __restrict__ WoT,
    const float* __restrict__ bo, float* __restrict__ out) {
  const int tid = threadIdx.x;
  const int wave = tid >> 5, lane = tid & 31;
  const int m_base = blockIdx.x * 128 + (wave >> 2) * 64;
  const int n_base = blockIdx.y * 128 + (wave & 3) * 32;

  v8f c[4][2];
  #pragma unroll
  for (int t = 0; t < 4; ++t) { c[t][0] = zero8(); c[t][1] = zero8(); }

  #pragma unroll
  for (int ks = 0; ks < INNER; ks += 32) {
    const v16h b0 = load_bt16(WoT, INNER, n_base, ks, lane);
    const v16h b1 = load_bt16(WoT, INNER, n_base + 16, ks, lane);
    #pragma unroll
    for (int t = 0; t < 4; ++t) {
      const v16h a = load_a16(go, INNER, m_base + 16 * t, ks, lane);
      c[t][0] = wmma32(a, b0, c[t][0]);
      c[t][1] = wmma32(a, b1, c[t][1]);
    }
  }

  #pragma unroll
  for (int u = 0; u < 2; ++u) {
    const int n = n_base + u * 16 + (lane & 15);
    const float bias = bo[n];
    #pragma unroll
    for (int t = 0; t < 4; ++t) {
      const int mrow = m_base + 16 * t + ((lane >> 4) << 3);
      const float* cf = (const float*)&c[t][u];
      #pragma unroll
      for (int r = 0; r < 8; ++r)
        out[(size_t)(mrow + r) * D_FEAT + n] = cf[r] + bias;
    }
  }
}

extern "C" void kernel_launch(void* const* d_in, const int* in_sizes, int n_in,
                              void* d_out, int out_size, void* d_ws, size_t ws_size,
                              hipStream_t stream) {
  (void)in_sizes; (void)n_in; (void)out_size; (void)ws_size;
  const float* x    = (const float*)d_in[0];
  const float* ln_g = (const float*)d_in[1];
  const float* ln_b = (const float*)d_in[2];
  const float* Wq   = (const float*)d_in[3];
  const float* Wkv  = (const float*)d_in[4];
  const float* Wg   = (const float*)d_in[5];
  const float* bg   = (const float*)d_in[6];
  const float* Wo   = (const float*)d_in[7];
  const float* bo   = (const float*)d_in[8];
  float* out = (float*)d_out;

  half_t* ws = (half_t*)d_ws;
  size_t off = 0;
  half_t* Wt  = ws + off; off += (size_t)2048 * 256;   // [Wq|Wk|Wv|Wg]^T f16
  half_t* WoT = ws + off; off += (size_t)256 * 512;    // Wo^T f16
  half_t* xn  = ws + off; off += (size_t)T_TOK * D_FEAT;
  half_t* qb  = ws + off; off += (size_t)T_TOK * INNER;
  half_t* kb  = ws + off; off += (size_t)T_TOK * INNER;
  half_t* vb  = ws + off; off += (size_t)T_TOK * INNER;
  half_t* gb  = ws + off; off += (size_t)T_TOK * INNER;
  half_t* gob = ws + off; off += (size_t)T_TOK * INNER;

  cvt_w_kernel<<<512, 256, 0, stream>>>(Wq, 512, 256, 512, Wt);
  cvt_w_kernel<<<512, 256, 0, stream>>>(Wkv, 1024, 256, 512, Wt + (size_t)512 * 256);
  cvt_w_kernel<<<512, 256, 0, stream>>>(Wkv + 512, 1024, 256, 512, Wt + (size_t)1024 * 256);
  cvt_w_kernel<<<512, 256, 0, stream>>>(Wg, 512, 256, 512, Wt + (size_t)1536 * 256);
  cvt_w_kernel<<<512, 256, 0, stream>>>(Wo, 256, 512, 256, WoT);

  ln_kernel<<<T_TOK / 8, 256, 0, stream>>>(x, ln_g, ln_b, xn);

  proj_kernel<<<dim3(T_TOK / 128, 2048 / 128), 256, 0, stream>>>(xn, Wt, bg, qb, kb,
                                                                 vb, gb);

  attn_kernel<<<N_ROWS * HEADS_N, 256, 131072, stream>>>(qb, kb, vb, gb, gob);

  outproj_kernel<<<dim3(T_TOK / 128, D_FEAT / 128), 256, 0, stream>>>(gob, WoT, bo,
                                                                      out);
}
